// GeometricBERT_86053964743042
// MI455X (gfx1250) — compile-verified, matrix operand order checked by reading, not run on hardware
//
#include <hip/hip_runtime.h>
#include <hip/hip_bf16.h>
#include <math.h>

#define D_  768
#define NL_ 12
#define NH_ 12
#define HD_ 64
#define FF_ 3072
#define B_  8
#define S_  512
#define R_  64
#define M_  (B_*S_)

typedef __attribute__((ext_vector_type(16))) __bf16 v16bf;
typedef __attribute__((ext_vector_type(8)))  float  v8f;

__device__ __forceinline__ v8f zero8() {
  v8f z;
#pragma unroll
  for (int i = 0; i < 8; ++i) z[i] = 0.0f;
  return z;
}

// Load a 16x32 bf16 fragment (A-layout per CDNA5 ISA 7.12.2) from an LDS tile
// stored row-major with K contiguous. Also used for B fragments by staging B
// tiles transposed (N-major, K contiguous).
__device__ __forceinline__ v16bf ld_frag(const __bf16* base, int stride) {
  int lane = threadIdx.x & 31;
  const __bf16* p = base + (lane & 15) * stride + ((lane >> 4) << 3);
  v16bf f;
  ((float4*)&f)[0] = *(const float4*)(p);        // halfs 0-7  : K = kb..kb+7
  ((float4*)&f)[1] = *(const float4*)(p + 16);   // halfs 8-15 : K = kb+16..kb+23
  return f;
}

__device__ __forceinline__ v8f wmma_bf16(v16bf a, v16bf b, v8f c) {
  return __builtin_amdgcn_wmma_f32_16x16x32_bf16(false, a, false, b, (short)0, c,
                                                 false, false);
}

// CDNA5 async global->LDS copy (ASYNCcnt-tracked, no VGPR round trip).
// Copies 64 contiguous bytes: 4 x GLOBAL_LOAD_ASYNC_TO_LDS_B128.
__device__ __forceinline__ void async_copy64B(void* lds_dst,
                                              const void* gbase,
                                              unsigned byte_off) {
  unsigned loff = (unsigned)(uintptr_t)lds_dst;
  asm volatile(
      "global_load_async_to_lds_b128 %0, %1, %2 offset:0\n\t"
      "global_load_async_to_lds_b128 %0, %1, %2 offset:16\n\t"
      "global_load_async_to_lds_b128 %0, %1, %2 offset:32\n\t"
      "global_load_async_to_lds_b128 %0, %1, %2 offset:48"
      :: "v"(loff), "v"(byte_off), "s"(gbase)
      : "memory");
}

__device__ __forceinline__ void wait_async() {
  asm volatile("s_wait_asynccnt 0" ::: "memory");
}

__device__ __forceinline__ void copy32h(__bf16* dst, const __bf16* src) {
  ((float4*)dst)[0] = ((const float4*)src)[0];
  ((float4*)dst)[1] = ((const float4*)src)[1];
  ((float4*)dst)[2] = ((const float4*)src)[2];
  ((float4*)dst)[3] = ((const float4*)src)[3];
}

__device__ __forceinline__ float block_reduce_sum256(float v, float* red) {
  int tid = threadIdx.x;
  red[tid] = v;
  __syncthreads();
  for (int off = 128; off > 0; off >>= 1) {
    if (tid < off) red[tid] += red[tid + off];
    __syncthreads();
  }
  float r = red[0];
  __syncthreads();
  return r;
}

// ---------------------------------------------------------------- embeddings
__global__ __launch_bounds__(256) void embed_kernel(
    const int* __restrict__ ids, const int* __restrict__ tts,
    const float* __restrict__ we, const float* __restrict__ pe,
    const float* __restrict__ te, const float* __restrict__ w,
    const float* __restrict__ b, float* __restrict__ h32,
    __bf16* __restrict__ h16) {
  __shared__ float red[256];
  int tok = blockIdx.x;
  int s = tok % S_;
  int id = ids[tok], tt = tts[tok];
  int tid = threadIdx.x;
  float v[3];
#pragma unroll
  for (int i = 0; i < 3; ++i) {
    int d = tid + i * 256;
    v[i] = we[(size_t)id * D_ + d] + pe[(size_t)s * D_ + d] + te[(size_t)tt * D_ + d];
  }
  float mean = block_reduce_sum256(v[0] + v[1] + v[2], red) * (1.0f / D_);
  float sq = 0.f;
#pragma unroll
  for (int i = 0; i < 3; ++i) { float c = v[i] - mean; sq += c * c; }
  float var = block_reduce_sum256(sq, red) * (1.0f / D_);
  float inv = rsqrtf(var + 1e-12f);
#pragma unroll
  for (int i = 0; i < 3; ++i) {
    int d = tid + i * 256;
    float o = (v[i] - mean) * inv * w[d] + b[d];
    h32[(size_t)tok * D_ + d] = o;
    h16[(size_t)tok * D_ + d] = (__bf16)o;
  }
}

// ------------------------------------------------------------ residual + LN
__global__ __launch_bounds__(256) void resid_ln_kernel(
    const float* __restrict__ x, const float* __restrict__ y,
    const float* __restrict__ w, const float* __restrict__ b,
    float* __restrict__ o32, __bf16* __restrict__ o16) {
  __shared__ float red[256];
  int tok = blockIdx.x, tid = threadIdx.x;
  float v[3];
#pragma unroll
  for (int i = 0; i < 3; ++i) {
    int d = tid + i * 256;
    v[i] = x[(size_t)tok * D_ + d] + y[(size_t)tok * D_ + d];
  }
  float mean = block_reduce_sum256(v[0] + v[1] + v[2], red) * (1.0f / D_);
  float sq = 0.f;
#pragma unroll
  for (int i = 0; i < 3; ++i) { float c = v[i] - mean; sq += c * c; }
  float var = block_reduce_sum256(sq, red) * (1.0f / D_);
  float inv = rsqrtf(var + 1e-12f);
#pragma unroll
  for (int i = 0; i < 3; ++i) {
    int d = tid + i * 256;
    float o = (v[i] - mean) * inv * w[d] + b[d];
    o32[(size_t)tok * D_ + d] = o;
    o16[(size_t)tok * D_ + d] = (__bf16)o;
  }
}

// --------------------------------------------------------------- WMMA GEMM
// C[M,N] = act(A_bf16[M,K] @ B_f32[K,N] + bias). 128x128 block tile, 8 waves,
// each wave a 32x64 region = 2x4 wmma tiles. K-step 64 (two wmma k-steps),
// double-buffered LDS; A tile staged with CDNA5 async global->LDS copies,
// B tile staged with coalesced float4 reads + f32->bf16 convert (transposed).
template <bool GELU, bool OUT32>
__global__ __launch_bounds__(256) void gemm_kernel(
    const __bf16* __restrict__ A, const float* __restrict__ Bw,
    const float* __restrict__ bias, float* __restrict__ C32,
    __bf16* __restrict__ C16, int M, int N, int K) {
  __shared__ __bf16 As[2][128 * 72];  // [m][k], K contiguous, padded to 72
  __shared__ __bf16 Bs[2][128 * 72];  // [n][k], transposed during staging
  int tid = threadIdx.x;
  int lane = tid & 31, wid = tid >> 5;
  int wm = (wid & 3) * 32, wn = (wid >> 2) * 64;
  int rowBase = blockIdx.y * 128, colBase = blockIdx.x * 128;

  v8f acc[2][4];
#pragma unroll
  for (int m = 0; m < 2; ++m)
#pragma unroll
    for (int n = 0; n < 4; ++n) acc[m][n] = zero8();

  int nk = K >> 6;
  int ar = tid >> 1, ac = (tid & 1) << 5;  // A staging: 64B per thread

  // ---- prologue: stage tile 0 into buffer 0
  async_copy64B(&As[0][ar * 72 + ac], A,
                (unsigned)(((size_t)(rowBase + ar) * K + ac) * 2));
  {
    const float* srcB = Bw + colBase;
#pragma unroll
    for (int i = 0; i < 8; ++i) {
      int lin = tid + (i << 8);
      int k = lin >> 5;
      int n4 = (lin & 31) << 2;
      float4 f = *(const float4*)(srcB + (size_t)k * N + n4);
      __bf16* d = &Bs[0][k];
      d[(n4 + 0) * 72] = (__bf16)f.x;
      d[(n4 + 1) * 72] = (__bf16)f.y;
      d[(n4 + 2) * 72] = (__bf16)f.z;
      d[(n4 + 3) * 72] = (__bf16)f.w;
    }
  }
  wait_async();
  __syncthreads();

  for (int kt = 0; kt < nk; ++kt) {
    int cur = kt & 1, nxt = cur ^ 1;
    if (kt + 1 < nk) {  // stage next tile while computing current
      async_copy64B(&As[nxt][ar * 72 + ac], A,
                    (unsigned)(((size_t)(rowBase + ar) * K + ((kt + 1) << 6) + ac) * 2));
      const float* srcB = Bw + (size_t)((kt + 1) << 6) * N + colBase;
#pragma unroll
      for (int i = 0; i < 8; ++i) {
        int lin = tid + (i << 8);
        int k = lin >> 5;
        int n4 = (lin & 31) << 2;
        float4 f = *(const float4*)(srcB + (size_t)k * N + n4);
        __bf16* d = &Bs[nxt][k];
        d[(n4 + 0) * 72] = (__bf16)f.x;
        d[(n4 + 1) * 72] = (__bf16)f.y;
        d[(n4 + 2) * 72] = (__bf16)f.z;
        d[(n4 + 3) * 72] = (__bf16)f.w;
      }
    }
#pragma unroll
    for (int ks = 0; ks < 64; ks += 32) {
      const __bf16* Ab = &As[cur][wm * 72 + ks];
      const __bf16* Bb = &Bs[cur][wn * 72 + ks];
      v16bf a0 = ld_frag(Ab, 72);
      v16bf a1 = ld_frag(Ab + 16 * 72, 72);
#pragma unroll
      for (int nt = 0; nt < 4; ++nt) {
        v16bf bfr = ld_frag(Bb + nt * 16 * 72, 72);
        acc[0][nt] = wmma_bf16(a0, bfr, acc[0][nt]);
        acc[1][nt] = wmma_bf16(a1, bfr, acc[1][nt]);
      }
    }
    wait_async();
    __syncthreads();
  }

  int hi = (lane >> 4) << 3;
#pragma unroll
  for (int mt = 0; mt < 2; ++mt)
#pragma unroll
    for (int nt = 0; nt < 4; ++nt)
#pragma unroll
      for (int v = 0; v < 8; ++v) {
        int gm = rowBase + wm + mt * 16 + v + hi;
        int gn = colBase + wn + nt * 16 + (lane & 15);
        float val = acc[mt][nt][v] + bias[gn];
        if (GELU) val = 0.5f * val * (1.0f + erff(val * 0.70710678118f));
        if (OUT32) C32[(size_t)gm * N + gn] = val;
        else       C16[(size_t)gm * N + gn] = (__bf16)val;
      }
}

// ----------------------------------------------------- mean pool over seq
__global__ __launch_bounds__(256) void meanpool_kernel(const float* __restrict__ h,
                                                       float* __restrict__ hm) {
  int b = blockIdx.x, tid = threadIdx.x;
#pragma unroll
  for (int i = 0; i < 3; ++i) {
    int d = tid + i * 256;
    float s = 0.f;
    for (int t = 0; t < S_; ++t) s += h[((size_t)b * S_ + t) * D_ + d];
    hm[(size_t)b * D_ + d] = s * (1.0f / S_);
  }
}

// --------------------------------- ctx projection (first HD cols) + sigmoid
__global__ __launch_bounds__(64) void ctxf_kernel(const float* __restrict__ hm,
                                                  const float* __restrict__ Wcp,
                                                  const float* __restrict__ bcp,
                                                  float* __restrict__ cw) {
  int b = blockIdx.x, n = threadIdx.x;
  float acc = bcp[n];
  for (int k = 0; k < D_; ++k) acc += hm[(size_t)b * D_ + k] * Wcp[(size_t)k * D_ + n];
  cw[b * HD_ + n] = 1.0f / (1.0f + expf(-acc));
}

// ------------------- metric = (Lm*cw)(Lm*cw)^T + diag(dg+eps) + 0.1I, invert
// Column-parallel Gauss-Jordan in LDS (SPD + 0.1 regularizer -> stable pivots).
__global__ __launch_bounds__(128) void metric_inv_kernel(
    const float* __restrict__ cw, const float* __restrict__ Lm,
    const float* __restrict__ dg, __bf16* __restrict__ ginvT) {
  __shared__ float lmod[64 * 64];
  __shared__ float aug[64 * 129];
  __shared__ float pc[64];
  int b = blockIdx.x, tid = threadIdx.x;
  for (int idx = tid; idx < 4096; idx += 128) {
    int d = idx >> 6, r = idx & 63;
    lmod[idx] = Lm[(size_t)d * R_ + r] * cw[b * HD_ + d];
  }
  __syncthreads();
  for (int idx = tid; idx < 4096; idx += 128) {
    int d = idx >> 6, e = idx & 63;
    float s = 0.f;
    for (int r = 0; r < 64; ++r) s += lmod[(d << 6) + r] * lmod[(e << 6) + r];
    if (d == e) s += dg[d] + 1e-6f + 0.1f;
    aug[d * 129 + e] = s;
    aug[d * 129 + 64 + e] = (d == e) ? 1.0f : 0.0f;
  }
  int j = tid;  // each thread owns augmented column j (0..127)
  for (int p = 0; p < 64; ++p) {
    __syncthreads();
    if (tid < 64) pc[tid] = aug[tid * 129 + p];  // snapshot pivot column
    __syncthreads();
    float prow = aug[p * 129 + j] / pc[p];
    aug[p * 129 + j] = prow;
    for (int r = 0; r < 64; ++r)
      if (r != p) aug[r * 129 + j] -= pc[r] * prow;
  }
  __syncthreads();
  for (int idx = tid; idx < 4096; idx += 128) {
    int d = idx >> 6, e = idx & 63;
    ginvT[(size_t)b * 4096 + e * 64 + d] = (__bf16)aug[d * 129 + 64 + e];
  }
}

// ----------------------- fused flash attention with metric: (q g^-1 k^T)/8
// Block = 128 threads (4 waves). One 64-row q tile per (qt, head, batch).
// g^-1, Q and K tiles staged with CDNA5 async global->LDS copies.
__global__ __launch_bounds__(128) void attn_kernel(
    const __bf16* __restrict__ Q, const __bf16* __restrict__ Kg,
    const __bf16* __restrict__ Vg, const __bf16* __restrict__ ginvT,
    float* __restrict__ ctx) {
  __shared__ __bf16 qg_s[64 * 72];
  __shared__ __bf16 gT_s[64 * 72];
  __shared__ __bf16 kt_s[64 * 72];  // also reused for Q staging
  __shared__ __bf16 vt_s[64 * 72];  // V^T tile: [e][kk]
  __shared__ __bf16 p_s[64 * 72];
  __shared__ float s_s[64 * 66];
  __shared__ float mrow[64], lrow[64], frow[64];

  int qt = blockIdx.x, hh = blockIdx.y, b = blockIdx.z;
  int tid = threadIdx.x, lane = tid & 31, wid = tid >> 5;
  int hi = (lane >> 4) << 3;
  int r = tid >> 1, c0 = (tid & 1) << 5;

  // stage g^-1 (transposed, N-major) and Q tile via async copies
  async_copy64B(gT_s + r * 72 + c0, ginvT,
                (unsigned)(((size_t)b * 4096 + r * 64 + c0) * 2));
  async_copy64B(kt_s + r * 72 + c0, Q,
                (unsigned)((((size_t)(b * S_ + qt * 64 + r)) * D_ + hh * HD_ + c0) * 2));
  wait_async();
  __syncthreads();

  // qg = (Q @ g^-1) * 1/sqrt(HD)
  v8f accq[4];
#pragma unroll
  for (int n = 0; n < 4; ++n) accq[n] = zero8();
#pragma unroll
  for (int ks = 0; ks < 64; ks += 32) {
    v16bf a = ld_frag(kt_s + wid * 16 * 72 + ks, 72);
#pragma unroll
    for (int nt = 0; nt < 4; ++nt)
      accq[nt] = wmma_bf16(a, ld_frag(gT_s + nt * 16 * 72 + ks, 72), accq[nt]);
  }
#pragma unroll
  for (int nt = 0; nt < 4; ++nt)
#pragma unroll
    for (int v = 0; v < 8; ++v) {
      int row = wid * 16 + v + hi;
      int col = nt * 16 + (lane & 15);
      qg_s[row * 72 + col] = (__bf16)(accq[nt][v] * 0.125f);
    }
  if (tid < 64) { mrow[tid] = -1e30f; lrow[tid] = 0.0f; }

  v8f acco[4];
#pragma unroll
  for (int n = 0; n < 4; ++n) acco[n] = zero8();
  __syncthreads();

  for (int kt = 0; kt < S_ / 64; ++kt) {
    {  // stage K tile [kk][e] (async) and V^T tile [e][kk] (scalar transpose)
      async_copy64B(kt_s + r * 72 + c0, Kg,
                    (unsigned)((((size_t)(b * S_ + kt * 64 + r)) * D_ + hh * HD_ + c0) * 2));
      alignas(16) __bf16 tmp[32];
      copy32h(tmp, Vg + ((size_t)(b * S_ + kt * 64 + r)) * D_ + hh * HD_ + c0);
#pragma unroll
      for (int i = 0; i < 32; ++i) vt_s[(c0 + i) * 72 + r] = tmp[i];
      wait_async();
    }
    __syncthreads();

    // S = qg @ K^T
    v8f accs[4];
#pragma unroll
    for (int n = 0; n < 4; ++n) accs[n] = zero8();
#pragma unroll
    for (int ks = 0; ks < 64; ks += 32) {
      v16bf a = ld_frag(qg_s + wid * 16 * 72 + ks, 72);
#pragma unroll
      for (int nt = 0; nt < 4; ++nt)
        accs[nt] = wmma_bf16(a, ld_frag(kt_s + nt * 16 * 72 + ks, 72), accs[nt]);
    }
#pragma unroll
    for (int nt = 0; nt < 4; ++nt)
#pragma unroll
      for (int v = 0; v < 8; ++v) {
        int row = wid * 16 + v + hi;
        s_s[row * 66 + nt * 16 + (lane & 15)] = accs[nt][v];
      }
    __syncthreads();

    // online softmax, one thread per row
    if (tid < 64) {
      int rr = tid;
      float mx = mrow[rr];
      for (int c = 0; c < 64; ++c) mx = fmaxf(mx, s_s[rr * 66 + c]);
      float fac = __expf(mrow[rr] - mx);
      float sum = lrow[rr] * fac;
      for (int c = 0; c < 64; ++c) {
        float pp = __expf(s_s[rr * 66 + c] - mx);
        sum += pp;
        p_s[rr * 72 + c] = (__bf16)pp;
      }
      mrow[rr] = mx; lrow[rr] = sum; frow[rr] = fac;
    }
    __syncthreads();

    // rescale O and accumulate P @ V
#pragma unroll
    for (int nt = 0; nt < 4; ++nt)
#pragma unroll
      for (int v = 0; v < 8; ++v) acco[nt][v] *= frow[wid * 16 + v + hi];
#pragma unroll
    for (int ks = 0; ks < 64; ks += 32) {
      v16bf a = ld_frag(p_s + wid * 16 * 72 + ks, 72);
#pragma unroll
      for (int nt = 0; nt < 4; ++nt)
        acco[nt] = wmma_bf16(a, ld_frag(vt_s + nt * 16 * 72 + ks, 72), acco[nt]);
    }
    __syncthreads();
  }

  if (tid < 64) frow[tid] = 1.0f / lrow[tid];
  __syncthreads();
#pragma unroll
  for (int nt = 0; nt < 4; ++nt)
#pragma unroll
    for (int v = 0; v < 8; ++v) {
      int row = wid * 16 + v + hi;
      int col = nt * 16 + (lane & 15);
      ctx[((size_t)(b * S_ + qt * 64 + row)) * D_ + hh * HD_ + col] =
          acco[nt][v] * frow[row];
    }
}

// ------------------------------------------------------------------- driver
extern "C" void kernel_launch(void* const* d_in, const int* in_sizes, int n_in,
                              void* d_out, int out_size, void* d_ws,
                              size_t ws_size, hipStream_t stream) {
  (void)in_sizes; (void)n_in; (void)out_size; (void)ws_size;
  const int*   ids = (const int*)d_in[0];
  const int*   tts = (const int*)d_in[1];
  const float* we  = (const float*)d_in[2];
  const float* pe  = (const float*)d_in[3];
  const float* te  = (const float*)d_in[4];
  const float* lnw = (const float*)d_in[5];
  const float* lnb = (const float*)d_in[6];
  const float* Wq  = (const float*)d_in[7];  const float* bq  = (const float*)d_in[8];
  const float* Wk  = (const float*)d_in[9];  const float* bk  = (const float*)d_in[10];
  const float* Wv  = (const float*)d_in[11]; const float* bv  = (const float*)d_in[12];
  const float* Lm  = (const float*)d_in[13]; const float* dgp = (const float*)d_in[14];
  const float* Wcp = (const float*)d_in[15]; const float* bcp = (const float*)d_in[16];
  const float* Wi  = (const float*)d_in[17]; const float* bi  = (const float*)d_in[18];
  const float* Wo  = (const float*)d_in[19]; const float* bo  = (const float*)d_in[20];
  const float* l1w = (const float*)d_in[21]; const float* l1b = (const float*)d_in[22];
  const float* l2w = (const float*)d_in[23]; const float* l2b = (const float*)d_in[24];

  char* p = (char*)d_ws;
  auto alloc = [&](size_t bytes) -> void* {
    void* r = (void*)p;
    p += (bytes + 255) & ~(size_t)255;
    return r;
  };
  float*  h32   = (float*)alloc((size_t)M_ * D_ * 4);
  __bf16* hb    = (__bf16*)alloc((size_t)M_ * D_ * 2);
  float*  a32   = (float*)alloc((size_t)M_ * D_ * 4);
  __bf16* ab    = (__bf16*)alloc((size_t)M_ * D_ * 2);
  float*  ctx32 = (float*)alloc((size_t)M_ * D_ * 4);
  float*  o32   = (float*)alloc((size_t)M_ * D_ * 4);
  __bf16* q16   = (__bf16*)alloc((size_t)M_ * D_ * 2);
  __bf16* k16   = (__bf16*)alloc((size_t)M_ * D_ * 2);
  __bf16* v16   = (__bf16*)alloc((size_t)M_ * D_ * 2);
  __bf16* f1    = (__bf16*)alloc((size_t)M_ * FF_ * 2);
  float*  hm    = (float*)alloc((size_t)B_ * D_ * 4);
  float*  cwb   = (float*)alloc((size_t)B_ * HD_ * 4);
  __bf16* gT    = (__bf16*)alloc((size_t)B_ * HD_ * HD_ * 2);

  embed_kernel<<<M_, 256, 0, stream>>>(ids, tts, we, pe, te, lnw, lnb, h32, hb);

  dim3 gD(D_ / 128, M_ / 128);
  dim3 gF(FF_ / 128, M_ / 128);
  for (int l = 0; l < NL_; ++l) {
    gemm_kernel<false, false><<<gD, 256, 0, stream>>>(
        hb, Wq + (size_t)l * D_ * D_, bq + (size_t)l * D_, nullptr, q16, M_, D_, D_);
    gemm_kernel<false, false><<<gD, 256, 0, stream>>>(
        hb, Wk + (size_t)l * D_ * D_, bk + (size_t)l * D_, nullptr, k16, M_, D_, D_);
    gemm_kernel<false, false><<<gD, 256, 0, stream>>>(
        hb, Wv + (size_t)l * D_ * D_, bv + (size_t)l * D_, nullptr, v16, M_, D_, D_);
    meanpool_kernel<<<B_, 256, 0, stream>>>(h32, hm);
    ctxf_kernel<<<B_, 64, 0, stream>>>(hm, Wcp + (size_t)l * D_ * D_,
                                       bcp + (size_t)l * D_, cwb);
    metric_inv_kernel<<<B_, 128, 0, stream>>>(cwb, Lm + (size_t)l * D_ * R_,
                                              dgp + (size_t)l * D_, gT);
    attn_kernel<<<dim3(S_ / 64, NH_, B_), 128, 0, stream>>>(q16, k16, v16, gT,
                                                            ctx32);
    resid_ln_kernel<<<M_, 256, 0, stream>>>(ctx32, h32, l1w + (size_t)l * D_,
                                            l1b + (size_t)l * D_, a32, ab);
    gemm_kernel<true, false><<<gF, 256, 0, stream>>>(
        ab, Wi + (size_t)l * D_ * FF_, bi + (size_t)l * FF_, nullptr, f1, M_, FF_, D_);
    gemm_kernel<false, true><<<gD, 256, 0, stream>>>(
        f1, Wo + (size_t)l * FF_ * D_, bo + (size_t)l * D_, o32, nullptr, M_, D_, FF_);
    float* outp = (l == NL_ - 1) ? (float*)d_out : h32;
    resid_ln_kernel<<<M_, 256, 0, stream>>>(o32, a32, l2w + (size_t)l * D_,
                                            l2b + (size_t)l * D_, outp, hb);
  }
}
